// VQVAE_84413287235780
// MI455X (gfx1250) — compile-verified, matrix operand order checked by reading, not run on hardware
//
#include <hip/hip_runtime.h>
#include <cstdint>
#include <cstddef>

// ---------------- CDNA5 WMMA types ----------------
typedef __bf16 bf16_t;
typedef bf16_t v16bf __attribute__((ext_vector_type(16)));
typedef float  v8f   __attribute__((ext_vector_type(8)));

#define ACT_NONE 0
#define ACT_RELU 1
#define ACT_GELU 2

__device__ __forceinline__ float act_fn(float v, int act) {
  if (act == ACT_RELU) return fmaxf(v, 0.0f);
  if (act == ACT_GELU) {
    // gelu(tanh approx), branch-free: tanh(u) = 1 - 2/(1+exp(2u))
    // v_exp_f32 + v_rcp_f32, no precise-division fixup chain
    const float c = 0.7978845608028654f;  // sqrt(2/pi)
    float u = c * (v + 0.044715f * v * v * v);
    float t = 1.0f - 2.0f * __builtin_amdgcn_rcpf(1.0f + __expf(2.0f * u));
    return 0.5f * v * (1.0f + t);
  }
  return v;
}

// =====================================================================
// Y[N,Cout] = act( X[N,K] @ W[K,Cout] + bias )     (row-major, f32 in/out)
// One wave32 computes a 16x32 tile of Y: one shared A fragment, two B
// fragments, two v_wmma_f32_16x16x32_bf16 per K-step.
// REQUIRES: N % 16 == 0, Cout % 32 == 0, K % 32 == 0 (no guards -> clean
// vectorized loads; other shapes are routed to the FMA kernel).
// Fragment layouts per CDNA5 ISA 7.12.2:
//   A (16x32 bf16): row = lane&15 ; elem j -> K = (j&8?16:0) + (lane>>4)*8 + (j&7)
//   B (32x16 bf16): col = lane&15 ; same K mapping
//   C/D (16x16 f32): col = lane&15 ; vgpr v -> row = v + 8*(lane>>4)
// =====================================================================
__global__ __launch_bounds__(256) void k_wmma_gemm(
    const float* __restrict__ X, const float* __restrict__ W,
    const float* __restrict__ bias, float* __restrict__ Y,
    int N, int K, int Cout, int act)
{
  const int wave = threadIdx.x >> 5;
  const int lane = threadIdx.x & 31;
  const int m0 = (blockIdx.x * 8 + wave) * 16;
  const int n0 = blockIdx.y * 32;
  if (m0 >= N) return;   // wave-uniform: EXEC stays all-ones for WMMA

  const int half = lane >> 4;
  const int l15  = lane & 15;
  // A row pointer, pre-offset by this half-wave's K phase
  const float* xr  = X + (size_t)(m0 + l15) * K + half * 8;
  // B column pointers (lane = column -> coalesced across the half-wave)
  const float* wc0 = W + n0 + l15;
  const float* wc1 = W + n0 + 16 + l15;

  v8f acc0 = {0.f, 0.f, 0.f, 0.f, 0.f, 0.f, 0.f, 0.f};
  v8f acc1 = {0.f, 0.f, 0.f, 0.f, 0.f, 0.f, 0.f, 0.f};

  for (int k0 = 0; k0 < K; k0 += 32) {
    // ---- A fragment: two contiguous 8-float runs -> 4x b128 loads ----
    float4 a0 = *(const float4*)(xr + k0);
    float4 a1 = *(const float4*)(xr + k0 + 4);
    float4 a2 = *(const float4*)(xr + k0 + 16);
    float4 a3 = *(const float4*)(xr + k0 + 20);
    v16bf a;
    a[0] = (bf16_t)a0.x;  a[1] = (bf16_t)a0.y;  a[2] = (bf16_t)a0.z;  a[3] = (bf16_t)a0.w;
    a[4] = (bf16_t)a1.x;  a[5] = (bf16_t)a1.y;  a[6] = (bf16_t)a1.z;  a[7] = (bf16_t)a1.w;
    a[8] = (bf16_t)a2.x;  a[9] = (bf16_t)a2.y;  a[10] = (bf16_t)a2.z; a[11] = (bf16_t)a2.w;
    a[12] = (bf16_t)a3.x; a[13] = (bf16_t)a3.y; a[14] = (bf16_t)a3.z; a[15] = (bf16_t)a3.w;

    // ---- B fragments: unconditional strided loads, coalesced per half-wave
    v16bf b0, b1;
#pragma unroll
    for (int j = 0; j < 16; ++j) {
      const size_t kk = (size_t)(k0 + ((j & 8) ? 16 : 0) + half * 8 + (j & 7)) * Cout;
      b0[j] = (bf16_t)wc0[kk];
      b1[j] = (bf16_t)wc1[kk];
    }

    // 8 args: (neg_a, A, neg_b, B, c_mod, C, reuse_a, reuse_b)
    acc0 = __builtin_amdgcn_wmma_f32_16x16x32_bf16(false, a, false, b0,
                                                   (short)0, acc0, false, false);
    acc1 = __builtin_amdgcn_wmma_f32_16x16x32_bf16(false, a, false, b1,
                                                   (short)0, acc1, false, false);
  }

  const float bc0 = bias ? bias[n0 + l15] : 0.0f;
  const float bc1 = bias ? bias[n0 + 16 + l15] : 0.0f;
#pragma unroll
  for (int v = 0; v < 8; ++v) {
    const int row = m0 + v + 8 * half;
    float* yr = Y + (size_t)row * Cout + n0 + l15;
    yr[0]  = act_fn(acc0[v] + bc0, act);
    yr[16] = act_fn(acc1[v] + bc1, act);
  }
}

// generic FMA GEMM for shapes WMMA can't take (K=4 input conv, 2/4-wide heads)
__global__ void k_small_gemm(const float* __restrict__ X, const float* __restrict__ W,
                             const float* __restrict__ b, float* __restrict__ Y,
                             long total, int K, int Cout, int act)
{
  long i = (long)blockIdx.x * blockDim.x + threadIdx.x;
  if (i >= total) return;
  int n = (int)(i / Cout), c = (int)(i % Cout);
  float acc = b ? b[c] : 0.0f;
  const float* xr = X + (size_t)n * K;
  for (int k = 0; k < K; ++k) acc = fmaf(xr[k], W[(size_t)k * Cout + c], acc);
  Y[i] = act_fn(acc, act);
}

// out[n*C+c] = bias[c]  (gconv output init before scatter-accumulate)
__global__ void k_fill_bias(float* __restrict__ out, const float* __restrict__ bias,
                            long total, int C)
{
  long i = (long)blockIdx.x * blockDim.x + threadIdx.x;
  if (i >= total) return;
  out[i] = bias[(int)(i % C)];
}

// one wave per edge: if et[e]==t, out[dst[e],:] += y[src[e],:]
__global__ void k_edge_scatter(const float* __restrict__ y, const int* __restrict__ src,
                               const int* __restrict__ dst, const int* __restrict__ et,
                               float* __restrict__ out, int E, int C, int t)
{
  int e = blockIdx.x * (blockDim.x >> 5) + (threadIdx.x >> 5);
  int lane = threadIdx.x & 31;
  if (e >= E) return;
  if (et[e] != t) return;
  const float* ys = y + (size_t)src[e] * C;
  float* od = out + (size_t)dst[e] * C;
  for (int c = lane; c < C; c += 32)
    atomicAdd(&od[c], ys[c]);
}

__global__ void k_zero64(float* s) { s[threadIdx.x] = 0.0f; }

// stats[0..31]=sum per group, stats[32..63]=sumsq per group (32 groups)
__global__ void k_gn_stats(const float* __restrict__ x, float* __restrict__ stats,
                           long total, int C)
{
  __shared__ float ssum[32], ssq[32];
  if (threadIdx.x < 32) { ssum[threadIdx.x] = 0.f; ssq[threadIdx.x] = 0.f; }
  __syncthreads();
  long i = (long)blockIdx.x * blockDim.x + threadIdx.x;
  if (i < total) {
    float v = x[i];
    int c = (int)(i % C);
    int g = c / (C >> 5);
    atomicAdd(&ssum[g], v);           // ds_add_f32
    atomicAdd(&ssq[g], v * v);
  }
  __syncthreads();
  if (threadIdx.x < 32) {
    atomicAdd(&stats[threadIdx.x], ssum[threadIdx.x]);
    atomicAdd(&stats[32 + threadIdx.x], ssq[threadIdx.x]);
  }
}

__global__ void k_gn_apply(const float* __restrict__ x, const float* __restrict__ g,
                           const float* __restrict__ b, const float* __restrict__ stats,
                           float* __restrict__ out, long total, int C, float invcnt, int act)
{
  long i = (long)blockIdx.x * blockDim.x + threadIdx.x;
  if (i >= total) return;
  int c = (int)(i % C);
  int gi = c / (C >> 5);
  float m = stats[gi] * invcnt;
  float var = stats[32 + gi] * invcnt - m * m;
  float r = (x[i] - m) * rsqrtf(var + 1e-5f);
  out[i] = act_fn(r * g[c] + b[c], act);
}

// 8 contiguous children -> parent mean
__global__ void k_pool8(const float* __restrict__ x, float* __restrict__ out, long total, int C)
{
  long i = (long)blockIdx.x * blockDim.x + threadIdx.x;
  if (i >= total) return;
  int n = (int)(i / C), c = (int)(i % C);
  float s = 0.f;
#pragma unroll
  for (int j = 0; j < 8; ++j) s += x[(size_t)(n * 8 + j) * C + c];
  out[i] = s * 0.125f;
}

// parent -> 8 children replicate
__global__ void k_repeat8(const float* __restrict__ x, float* __restrict__ out, long total, int C)
{
  long i = (long)blockIdx.x * blockDim.x + threadIdx.x;
  if (i >= total) return;
  int n = (int)(i / C), c = (int)(i % C);
  out[i] = x[(size_t)(n >> 3) * C + c];
}

__global__ void k_add_inplace(float* __restrict__ a, const float* __restrict__ b, long total)
{
  long i = (long)blockIdx.x * blockDim.x + threadIdx.x;
  if (i < total) a[i] += b[i];
}

__global__ void k_copy(float* __restrict__ d, const float* __restrict__ s, long total)
{
  long i = (long)blockIdx.x * blockDim.x + threadIdx.x;
  if (i < total) d[i] = s[i];
}

// =====================================================================
// Host-side orchestration helpers (all launches on `stream`)
// =====================================================================
struct RB   { const float *g1,*be1,*W1,*b1,*g2,*be2,*W2,*b2; };
struct SAMP { const float *W,*b,*g,*be; };
struct HD   { const float *W1,*b1,*g,*be,*W2,*b2; };
struct EDG  { const int *src,*dst,*et; };

static inline int cdiv(long a, long b) { return (int)((a + b - 1) / b); }

static void gemm(hipStream_t s, const float* X, const float* W, const float* bias,
                 float* Y, int N, int K, int Cout, int act)
{
  if ((K % 32) == 0 && (Cout % 32) == 0) {
    dim3 g(cdiv(N / 16, 8), Cout / 32);
    k_wmma_gemm<<<g, 256, 0, s>>>(X, W, bias, Y, N, K, Cout, act);
  } else {
    long tot = (long)N * Cout;
    k_small_gemm<<<cdiv(tot, 256), 256, 0, s>>>(X, W, bias, Y, tot, K, Cout, act);
  }
}

static void gn_act(hipStream_t s, const float* x, const float* g, const float* b,
                   float* out, float* stats, int N, int C, int act)
{
  k_zero64<<<1, 64, 0, s>>>(stats);
  long tot = (long)N * C;
  int nb = cdiv(tot, 256);
  k_gn_stats<<<nb, 256, 0, s>>>(x, stats, tot, C);
  float invcnt = 1.0f / ((float)N * (float)(C >> 5));
  k_gn_apply<<<nb, 256, 0, s>>>(x, g, b, stats, out, tot, C, invcnt, act);
}

static void gconv(hipStream_t s, const float* x, const float* W, const float* bias,
                  EDG e, int N, int Cin, int Cout, float* out, float* ybuf)
{
  long tot = (long)N * Cout;
  k_fill_bias<<<cdiv(tot, 256), 256, 0, s>>>(out, bias, tot, Cout);
  int E = 7 * N;
  for (int t = 0; t < 7; ++t) {
    gemm(s, x, W + (size_t)t * Cin * Cout, nullptr, ybuf, N, Cin, Cout, ACT_NONE);
    k_edge_scatter<<<cdiv(E, 8), 256, 0, s>>>(ybuf, e.src, e.dst, e.et, out, E, Cout, t);
  }
}

static void resblk(hipStream_t s, float* x, const RB& p, EDG e, int N, int C, int act,
                   float* t1, float* t2, float* ybuf, float* stats)
{
  gn_act(s, x, p.g1, p.be1, t1, stats, N, C, act);
  gconv(s, t1, p.W1, p.b1, e, N, C, C, t2, ybuf);
  gn_act(s, t2, p.g2, p.be2, t1, stats, N, C, act);
  gconv(s, t1, p.W2, p.b2, e, N, C, C, t2, ybuf);
  k_add_inplace<<<cdiv((long)N * C, 256), 256, 0, s>>>(x, t2, (long)N * C);
}

static void down(hipStream_t s, float* x, const SAMP& p, int N, int Cin, int Cout, int act,
                 float* t1, float* t2, float* stats)
{
  int Np = N / 8;
  k_pool8<<<cdiv((long)Np * Cin, 256), 256, 0, s>>>(x, t1, (long)Np * Cin, Cin);
  gemm(s, t1, p.W, p.b, t2, Np, Cin, Cout, ACT_NONE);
  gn_act(s, t2, p.g, p.be, x, stats, Np, Cout, act);
}

static void up(hipStream_t s, float* x, const SAMP& p, int N, int Cin, int Cout, int act,
               float* t1, float* t2, float* stats)
{
  gemm(s, x, p.W, p.b, t1, N, Cin, Cout, ACT_NONE);
  int Nc = N * 8;
  k_repeat8<<<cdiv((long)Nc * Cout, 256), 256, 0, s>>>(t1, t2, (long)Nc * Cout, Cout);
  gn_act(s, t2, p.g, p.be, x, stats, Nc, Cout, act);
}

static void head(hipStream_t s, const float* x, const HD& p, int N, int C, int Co, int act,
                 float* outp, float* t1, float* t2, float* stats)
{
  gemm(s, x, p.W1, p.b1, t1, N, C, 32, ACT_NONE);
  gn_act(s, t1, p.g, p.be, t2, stats, N, 32, act);
  long tot = (long)N * Co;
  k_small_gemm<<<cdiv(tot, 256), 256, 0, s>>>(t2, p.W2, p.b2, outp, tot, 32, Co, ACT_NONE);
}

// =====================================================================
extern "C" void kernel_launch(void* const* d_in, const int* in_sizes, int n_in,
                              void* d_out, int out_size, void* d_ws, size_t ws_size,
                              hipStream_t stream)
{
  (void)in_sizes; (void)n_in; (void)out_size; (void)ws_size;
  const int N6 = 262144, N5 = 32768, N4 = 4096, N3 = 512, N2 = 64;

  const float* data = (const float*)d_in[0];
  EDG e6 = {(const int*)d_in[1],  (const int*)d_in[2],  (const int*)d_in[3]};
  EDG e5 = {(const int*)d_in[4],  (const int*)d_in[5],  (const int*)d_in[6]};
  EDG e4 = {(const int*)d_in[7],  (const int*)d_in[8],  (const int*)d_in[9]};
  EDG e3 = {(const int*)d_in[10], (const int*)d_in[11], (const int*)d_in[12]};
  EDG e2 = {(const int*)d_in[13], (const int*)d_in[14], (const int*)d_in[15]};

  // params flattened in setup_inputs() insertion order, leaves at d_in[16..]
  int pi = 16;
  auto P  = [&]() { return (const float*)d_in[pi++]; };
  auto rb = [&]() { RB r;   r.g1=P(); r.be1=P(); r.W1=P(); r.b1=P();
                            r.g2=P(); r.be2=P(); r.W2=P(); r.b2=P(); return r; };
  auto sp = [&]() { SAMP r; r.W=P(); r.b=P(); r.g=P(); r.be=P(); return r; };
  auto hd = [&]() { HD r;   r.W1=P(); r.b1=P(); r.g=P(); r.be=P(); r.W2=P(); r.b2=P(); return r; };

  const float *c1W = P(), *c1b = P(), *c1g = P(), *c1be = P();
  RB erb6 = rb(); SAMP edn6 = sp(); RB erb5 = rb(); SAMP edn5 = sp(); RB erb4 = rb();
  const float *preW = P(), *preb = P(), *postW = P(), *postb = P();
  RB uerb4 = rb(); SAMP udn4 = sp(); RB uerb3 = rb(); SAMP udn3 = sp(); RB uerb2 = rb();
  RB udrb2 = rb(); SAMP uup2 = sp(); RB udrb3 = rb(); SAMP uup3 = sp(); RB udrb4 = rb();
  RB drb4 = rb(); HD pred4 = hd(); HD reg4 = hd(); SAMP dup4 = sp();
  RB drb5 = rb(); HD pred5 = hd(); HD reg5 = hd(); SAMP dup5 = sp();
  RB drb6 = rb(); HD pred6 = hd(); HD reg6 = hd();

  // workspace layout (floats): 4 big buffers of N6*32 + skips + stats
  const size_t BIG = (size_t)N6 * 32;
  float* ws = (float*)d_ws;
  float* BA = ws;                    // current activation
  float* BB = ws + BIG;              // temp 1
  float* BC = ws + 2 * BIG;          // temp 2
  float* YB = ws + 3 * BIG;          // per-edge-type GEMM output
  float* U4 = ws + 4 * BIG;          // skip [N4,64]
  float* U3 = U4 + (size_t)N4 * 64;  // skip [N3,128]
  float* ST = U3 + (size_t)N3 * 128; // 64 floats of GN stats

  // output layout: (log4, log5, log6, sig4, sig5, sig6) flat
  float* out  = (float*)d_out;
  float* log4 = out;
  float* log5 = log4 + (size_t)N4 * 2;
  float* log6 = log5 + (size_t)N5 * 2;
  float* sig4 = log6 + (size_t)N6 * 2;
  float* sig5 = sig4 + (size_t)N4 * 4;
  float* sig6 = sig5 + (size_t)N5 * 4;

  hipStream_t s = stream;

  // ---------------- Encoder (6 -> 4) ----------------
  gconv(s, data, c1W, c1b, e6, N6, 4, 32, BA, YB);
  gn_act(s, BA, c1g, c1be, BA, ST, N6, 32, ACT_RELU);
  resblk(s, BA, erb6, e6, N6, 32, ACT_RELU, BB, BC, YB, ST);
  down(s, BA, edn6, N6, 32, 32, ACT_RELU, BB, BC, ST);
  resblk(s, BA, erb5, e5, N5, 32, ACT_RELU, BB, BC, YB, ST);
  down(s, BA, edn5, N5, 32, 64, ACT_RELU, BB, BC, ST);
  resblk(s, BA, erb4, e4, N4, 64, ACT_RELU, BB, BC, YB, ST);
  gemm(s, BA, preW, preb, BB, N4, 64, 256, ACT_NONE);    // code
  gemm(s, BB, postW, postb, BA, N4, 256, 64, ACT_NONE);  // post_proj

  // ---------------- TinyUNet (4 -> 2 -> 4) ----------------
  resblk(s, BA, uerb4, e4, N4, 64, ACT_RELU, BB, BC, YB, ST);
  k_copy<<<cdiv((long)N4 * 64, 256), 256, 0, s>>>(U4, BA, (long)N4 * 64);
  down(s, BA, udn4, N4, 64, 128, ACT_RELU, BB, BC, ST);
  resblk(s, BA, uerb3, e3, N3, 128, ACT_RELU, BB, BC, YB, ST);
  k_copy<<<cdiv((long)N3 * 128, 256), 256, 0, s>>>(U3, BA, (long)N3 * 128);
  down(s, BA, udn3, N3, 128, 256, ACT_RELU, BB, BC, ST);
  resblk(s, BA, uerb2, e2, N2, 256, ACT_RELU, BB, BC, YB, ST);
  resblk(s, BA, udrb2, e2, N2, 256, ACT_RELU, BB, BC, YB, ST);
  up(s, BA, uup2, N2, 256, 128, ACT_RELU, BB, BC, ST);
  k_add_inplace<<<cdiv((long)N3 * 128, 256), 256, 0, s>>>(BA, U3, (long)N3 * 128);
  resblk(s, BA, udrb3, e3, N3, 128, ACT_RELU, BB, BC, YB, ST);
  up(s, BA, uup3, N3, 128, 64, ACT_RELU, BB, BC, ST);
  k_add_inplace<<<cdiv((long)N4 * 64, 256), 256, 0, s>>>(BA, U4, (long)N4 * 64);
  resblk(s, BA, udrb4, e4, N4, 64, ACT_RELU, BB, BC, YB, ST);

  // ---------------- DecodeOctree (4 -> 6, gelu) ----------------
  resblk(s, BA, drb4, e4, N4, 64, ACT_GELU, BB, BC, YB, ST);
  head(s, BA, pred4, N4, 64, 2, ACT_GELU, log4, BB, BC, ST);
  head(s, BA, reg4,  N4, 64, 4, ACT_GELU, sig4, BB, BC, ST);
  up(s, BA, dup4, N4, 64, 32, ACT_GELU, BB, BC, ST);
  resblk(s, BA, drb5, e5, N5, 32, ACT_GELU, BB, BC, YB, ST);
  head(s, BA, pred5, N5, 32, 2, ACT_GELU, log5, BB, BC, ST);
  head(s, BA, reg5,  N5, 32, 4, ACT_GELU, sig5, BB, BC, ST);
  up(s, BA, dup5, N5, 32, 32, ACT_GELU, BB, BC, ST);
  resblk(s, BA, drb6, e6, N6, 32, ACT_GELU, BB, BC, YB, ST);
  head(s, BA, pred6, N6, 32, 2, ACT_GELU, log6, BB, BC, ST);
  head(s, BA, reg6,  N6, 32, 4, ACT_GELU, sig6, BB, BC, ST);
}
